// Seq2Seq_d_59949153517883
// MI455X (gfx1250) — matrix operand order repaired and checked by reading, not compiled, not run on hardware
//
#include <hip/hip_runtime.h>
#include <hip/hip_bf16.h>

typedef __attribute__((ext_vector_type(16))) __bf16 v16bf;
typedef __attribute__((ext_vector_type(8)))  float  v8f;

// Sizes
// B=32, S=256, ENC=768, DH=128, EMB=200, V=32000, T=96
// fc K = 128+768+200 = 1096, padded to 1120 = 35*32
#define NKT   35
#define NTILE 2000   // 32000/16

// ---------------- one-time prep kernels ----------------

// Pack fc_W [32000,1096] f32 -> bf16 WMMA-B fragments.
// Fragment (ntile, kt): lane<16 -> n=ntile*16+lane, K=kt*32+0..15
//                       lane>=16 -> n=ntile*16+lane-16, K=kt*32+16..31
// Stored fragment-contiguous: pB[frag*512 + lane*16 + e]
__global__ void pack_fcW(const float* __restrict__ fcW, __bf16* __restrict__ pB) {
    int gid  = blockIdx.x * 256 + threadIdx.x;      // 70000*32 = 2,240,000 exactly
    int lane = gid & 31;
    int frag = gid >> 5;
    int ntile = frag / NKT;
    int kt    = frag % NKT;
    int n  = ntile * 16 + (lane & 15);
    int k0 = kt * 32 + ((lane >> 4) * 16);
    const float* src = fcW + (size_t)n * 1096;
    v16bf o;
#pragma unroll
    for (int e = 0; e < 16; ++e) {
        int k = k0 + e;
        float v = (k < 1096) ? src[k] : 0.0f;
        o[e] = (__bf16)v;
    }
    *(v16bf*)(pB + (size_t)frag * 512 + (size_t)lane * 16) = o;
}

// enc_proj[b,s,j] = attn_b[j] + sum_k enc[b,s,k] * attn_W[128+k, j]
__global__ void enc_proj_kernel(const float* __restrict__ enc,
                                const float* __restrict__ attn_W,
                                const float* __restrict__ attn_b,
                                float* __restrict__ ep) {
    int bs = blockIdx.x;           // 0..8191 (b*256+s)
    int j  = threadIdx.x;          // 0..127
    const float* e = enc + (size_t)bs * 768;
    float sum = attn_b[j];
    for (int k = 0; k < 768; ++k)
        sum = fmaf(e[k], attn_W[(size_t)(128 + k) * 128 + j], sum);
    ep[(size_t)bs * 128 + j] = sum;
}

// emb_all[si,b,:] = embedding[trg[b,si],:]   (si = 0..94, input tokens trg[:, :-1])
__global__ void gather_emb(const int* __restrict__ trg,
                           const float* __restrict__ embedding,
                           float* __restrict__ emb_all) {
    int blk = blockIdx.x;          // 95*32
    int si = blk / 32, b = blk % 32;
    int e = threadIdx.x;
    if (e >= 200) return;
    int tok = trg[b * 96 + si];
    emb_all[((size_t)(si * 32 + b)) * 200 + e] = embedding[(size_t)tok * 200 + e];
}

__global__ void zero_out_t0(float* __restrict__ out) {
    int gid = blockIdx.x * 256 + threadIdx.x;   // 32*32000 = 1,024,000 exactly
    int b = gid / 32000, v = gid % 32000;
    out[(size_t)b * 96 * 32000 + v] = 0.0f;
}

__global__ void zero_state(float* __restrict__ h, float* __restrict__ c) {
    int gid = blockIdx.x * 256 + threadIdx.x;
    if (gid < 32 * 128) { h[gid] = 0.0f; c[gid] = 0.0f; }
}

// ---------------- per-step kernels ----------------

// hproj[b,j] = sum_k h[b,k] * attn_W[k,j]   (k<128)
__global__ void hproj_kernel(const float* __restrict__ h,
                             const float* __restrict__ attn_W,
                             float* __restrict__ hp) {
    int b = blockIdx.x, j = threadIdx.x;
    const float* hb = h + b * 128;
    float sum = 0.0f;
    for (int k = 0; k < 128; ++k)
        sum = fmaf(hb[k], attn_W[(size_t)k * 128 + j], sum);
    hp[b * 128 + j] = sum;
}

// esc[b,s] = sum_j tanh(ep[b,s,j] + hproj[b,j]) * attn_v[j]  (one wave per (b,s))
__global__ void attn_scores(const float* __restrict__ ep,
                            const float* __restrict__ hp,
                            const float* __restrict__ attn_v,
                            float* __restrict__ esc) {
    int gid  = blockIdx.x * 256 + threadIdx.x;   // 1024 blocks -> 8192 waves
    int w    = gid >> 5;
    int lane = gid & 31;
    int b = w >> 8, s = w & 255;
    const float* epp = ep + ((size_t)(b * 256 + s)) * 128;
    const float* hpb = hp + b * 128;
    float sum = 0.0f;
#pragma unroll
    for (int jj = 0; jj < 4; ++jj) {
        int j = lane + jj * 32;
        sum += tanhf(epp[j] + hpb[j]) * attn_v[j];
    }
#pragma unroll
    for (int off = 16; off > 0; off >>= 1) sum += __shfl_down(sum, off, 32);
    if (lane == 0) esc[b * 256 + s] = sum;
}

// softmax over s then weighted[b,d] = sum_s a[s]*enc[b,s,d]
__global__ void softmax_weighted(const float* __restrict__ esc,
                                 const float* __restrict__ enc,
                                 float* __restrict__ weighted) {
    __shared__ float sa[256];
    __shared__ float red[256];
    int b = blockIdx.x, tid = threadIdx.x;
    float ev = esc[b * 256 + tid];
    red[tid] = ev; __syncthreads();
    for (int off = 128; off > 0; off >>= 1) {
        if (tid < off) red[tid] = fmaxf(red[tid], red[tid + off]);
        __syncthreads();
    }
    float mx = red[0]; __syncthreads();
    float ex = expf(ev - mx);
    red[tid] = ex; __syncthreads();
    for (int off = 128; off > 0; off >>= 1) {
        if (tid < off) red[tid] += red[tid + off];
        __syncthreads();
    }
    float inv = 1.0f / red[0];
    sa[tid] = ex * inv;
    __syncthreads();
    const float* eb = enc + (size_t)b * 256 * 768;
#pragma unroll
    for (int dd = 0; dd < 3; ++dd) {
        int d = tid + dd * 256;
        float acc = 0.0f;
        for (int s = 0; s < 256; ++s)
            acc = fmaf(sa[s], eb[(size_t)s * 768 + d], acc);
        weighted[b * 768 + d] = acc;
    }
}

// gates[b,j] = b_ih[j]+b_hh[j] + x@W_ih[j,:] + h@W_hh[j,:], x=cat(emb,weighted)
// lane = batch (weight loads wave-uniform)
__global__ void lstm_gates(const float* __restrict__ emb_all, int si,
                           const float* __restrict__ weighted,
                           const float* __restrict__ h,
                           const float* __restrict__ W_ih,
                           const float* __restrict__ W_hh,
                           const float* __restrict__ b_ih,
                           const float* __restrict__ b_hh,
                           float* __restrict__ gates) {
    int gid = blockIdx.x * 256 + threadIdx.x;    // 64 blocks -> 16384 = 512*32
    int j = gid >> 5, b = gid & 31;
    const float* wih = W_ih + (size_t)j * 968;
    const float* whh = W_hh + (size_t)j * 128;
    const float* eb  = emb_all + ((size_t)(si * 32 + b)) * 200;
    const float* wb  = weighted + b * 768;
    const float* hb  = h + b * 128;
    float sum = b_ih[j] + b_hh[j];
    for (int k = 0; k < 200; ++k) sum = fmaf(eb[k], wih[k], sum);
    for (int k = 0; k < 768; ++k) sum = fmaf(wb[k], wih[200 + k], sum);
    for (int k = 0; k < 128; ++k) sum = fmaf(hb[k], whh[k], sum);
    gates[b * 512 + j] = sum;
}

__global__ void lstm_update(const float* __restrict__ gates,
                            float* __restrict__ h, float* __restrict__ c) {
    int b = blockIdx.x, j = threadIdx.x;   // 32 x 128
    const float* g = gates + b * 512;
    float ig = 1.0f / (1.0f + expf(-g[j]));
    float fg = 1.0f / (1.0f + expf(-g[128 + j]));
    float gg = tanhf(g[256 + j]);
    float og = 1.0f / (1.0f + expf(-g[384 + j]));
    float cn = fg * c[b * 128 + j] + ig * gg;
    c[b * 128 + j] = cn;
    h[b * 128 + j] = og * tanhf(cn);
}

// Pack A fragments (x = cat(h, weighted, emb), M=batch=32 -> 2 mtiles, 35 ktiles)
// Contiguous k order (same as pB, not AMD's table order): lane<16 holds K {0..15}, lane>=16 holds {16..31}
__global__ void pack_A(const float* __restrict__ h,
                       const float* __restrict__ weighted,
                       const float* __restrict__ emb_all, int si,
                       __bf16* __restrict__ pA) {
    int bx = blockIdx.x;        // 0..69 = mtile*35 + kt
    int lane = threadIdx.x;     // 0..31
    int mtile = bx / NKT, kt = bx % NKT;
    int b = mtile * 16 + (lane & 15);
    int khalf = (lane >> 4) * 16;
    const float* eb = emb_all + ((size_t)(si * 32 + b)) * 200;
    v16bf o;
#pragma unroll
    for (int e = 0; e < 16; ++e) {
        int koff = e + khalf;
        int k = kt * 32 + koff;
        float v;
        if (k < 128)       v = h[b * 128 + k];
        else if (k < 896)  v = weighted[b * 768 + (k - 128)];
        else if (k < 1096) v = eb[k - 896];
        else               v = 0.0f;
        o[e] = (__bf16)v;
    }
    *(v16bf*)(pA + ((size_t)bx * 32 + lane) * 16) = o;
}

// Main GEMM: pred[32, 32000] = x[32,1120] @ fcW_bf16.T, bf16 WMMA, f32 acc.
// Each wave owns TWO adjacent 16-col vocab tiles -> 4 independent accumulators:
// 8 b128 loads per 4 WMMAs, and deep WMMA ILP to cover the RAW latency.
__global__ void __launch_bounds__(256)
fc_gemm(const __bf16* __restrict__ pA, const __bf16* __restrict__ pB,
        const float* __restrict__ fc_b, float* __restrict__ out, int t) {
    int tid  = threadIdx.x;
    int lane = tid & 31;
    int wave = tid >> 5;
    int ntile0 = (blockIdx.x * 8 + wave) * 2;   // 125 blocks * 8 waves * 2 = 2000 tiles
    const v16bf* A   = (const v16bf*)pA;
    const v16bf* B0p = (const v16bf*)pB + (size_t)(ntile0    ) * NKT * 32 + lane;
    const v16bf* B1p = (const v16bf*)pB + (size_t)(ntile0 + 1) * NKT * 32 + lane;
    v8f acc00 = {};   // mtile0 x ntile0
    v8f acc10 = {};   // mtile1 x ntile0
    v8f acc01 = {};   // mtile0 x ntile1
    v8f acc11 = {};   // mtile1 x ntile1
    for (int kt = 0; kt < NKT; ++kt) {
        v16bf a0 = A[(0 * NKT + kt) * 32 + lane];
        v16bf a1 = A[(1 * NKT + kt) * 32 + lane];
        v16bf b0 = B0p[(size_t)kt * 32];
        v16bf b1 = B1p[(size_t)kt * 32];
        __builtin_prefetch((const void*)(B0p + (size_t)(kt + 2) * 32), 0, 1);
        __builtin_prefetch((const void*)(B1p + (size_t)(kt + 2) * 32), 0, 1);
        acc00 = __builtin_amdgcn_wmma_f32_16x16x32_bf16(false, a0, false, b0,
                                                        (short)0, acc00, false, false);
        acc10 = __builtin_amdgcn_wmma_f32_16x16x32_bf16(false, a1, false, b0,
                                                        (short)0, acc10, false, false);
        acc01 = __builtin_amdgcn_wmma_f32_16x16x32_bf16(false, a0, false, b1,
                                                        (short)0, acc01, false, false);
        acc11 = __builtin_amdgcn_wmma_f32_16x16x32_bf16(false, a1, false, b1,
                                                        (short)0, acc11, false, false);
    }
    int mBase = (lane >> 4) * 8;
    // ntile0 columns
    {
        int n = ntile0 * 16 + (lane & 15);
        float bias = fc_b[n];
        float* ob = out + (size_t)t * 32000 + n;
#pragma unroll
        for (int r = 0; r < 8; ++r) {
            int m0 = mBase + r;
            ob[(size_t)m0 * 96 * 32000]        = acc00[r] + bias;   // batches 0..15
            ob[(size_t)(m0 + 16) * 96 * 32000] = acc10[r] + bias;   // batches 16..31
        }
    }
    // ntile1 columns
    {
        int n = (ntile0 + 1) * 16 + (lane & 15);
        float bias = fc_b[n];
        float* ob = out + (size_t)t * 32000 + n;
#pragma unroll
        for (int r = 0; r < 8; ++r) {
            int m0 = mBase + r;
            ob[(size_t)m0 * 96 * 32000]        = acc01[r] + bias;
            ob[(size_t)(m0 + 16) * 96 * 32000] = acc11[r] + bias;
        }
    }
}

// argmax over vocab, first-occurrence tie-break (jnp.argmax semantics)
__global__ void argmax_kernel(const float* __restrict__ out, float* __restrict__ pred) {
    __shared__ float bv[256];
    __shared__ int   bi[256];
    int blk = blockIdx.x;          // b*96 + t
    int tid = threadIdx.x;
    const float* row = out + (size_t)blk * 32000;
    float best = -3.402823466e38f;
    int bidx = 0;
    for (int v = tid; v < 32000; v += 256) {
        float x = row[v];
        if (x > best) { best = x; bidx = v; }
    }
    bv[tid] = best; bi[tid] = bidx; __syncthreads();
    for (int off = 128; off > 0; off >>= 1) {
        if (tid < off) {
            if (bv[tid + off] > bv[tid] ||
                (bv[tid + off] == bv[tid] && bi[tid + off] < bi[tid])) {
                bv[tid] = bv[tid + off]; bi[tid] = bi[tid + off];
            }
        }
        __syncthreads();
    }
    if (tid == 0) pred[blk] = (float)bi[0];
}

// ---------------- launch ----------------

extern "C" void kernel_launch(void* const* d_in, const int* in_sizes, int n_in,
                              void* d_out, int out_size, void* d_ws, size_t ws_size,
                              hipStream_t stream) {
    const float* enc       = (const float*)d_in[0];
    const int*   trg       = (const int*)  d_in[1];
    const float* embedding = (const float*)d_in[2];
    const float* attn_W    = (const float*)d_in[3];
    const float* attn_b    = (const float*)d_in[4];
    const float* attn_v    = (const float*)d_in[5];
    const float* W_ih      = (const float*)d_in[6];
    const float* W_hh      = (const float*)d_in[7];
    const float* b_ih      = (const float*)d_in[8];
    const float* b_hh      = (const float*)d_in[9];
    const float* fc_W      = (const float*)d_in[10];
    const float* fc_b      = (const float*)d_in[11];
    (void)in_sizes; (void)n_in; (void)out_size; (void)ws_size;

    float* out  = (float*)d_out;
    float* pred = out + (size_t)32 * 96 * 32000;

    char* ws = (char*)d_ws;
    size_t off = 0;
    auto alloc = [&](size_t bytes) -> void* {
        void* p = ws + off;
        off += (bytes + 255) & ~(size_t)255;
        return p;
    };
    __bf16* pB      = (__bf16*)alloc((size_t)NTILE * NKT * 512 * 2);  // ~71.7 MB
    float*  ep      = (float*) alloc((size_t)32 * 256 * 128 * 4);     // 4 MB
    float*  emb_all = (float*) alloc((size_t)95 * 32 * 200 * 4);      // 2.4 MB
    float*  hp      = (float*) alloc((size_t)32 * 128 * 4);
    float*  esc     = (float*) alloc((size_t)32 * 256 * 4);
    float*  wsum    = (float*) alloc((size_t)32 * 768 * 4);
    float*  gates   = (float*) alloc((size_t)32 * 512 * 4);
    float*  hs      = (float*) alloc((size_t)32 * 128 * 4);
    float*  cs      = (float*) alloc((size_t)32 * 128 * 4);
    __bf16* pA      = (__bf16*)alloc((size_t)70 * 512 * 2);

    // one-time prep (re-done every call: deterministic, no cross-call state)
    pack_fcW       <<<8750, 256, 0, stream>>>(fc_W, pB);
    enc_proj_kernel<<<8192, 128, 0, stream>>>(enc, attn_W, attn_b, ep);
    gather_emb     <<<95 * 32, 256, 0, stream>>>(trg, embedding, emb_all);
    zero_out_t0    <<<4000, 256, 0, stream>>>(out);
    zero_state     <<<16, 256, 0, stream>>>(hs, cs);

    for (int t = 1; t < 96; ++t) {
        int si = t - 1;
        hproj_kernel    <<<32,   128, 0, stream>>>(hs, attn_W, hp);
        attn_scores     <<<1024, 256, 0, stream>>>(ep, hp, attn_v, esc);
        softmax_weighted<<<32,   256, 0, stream>>>(esc, enc, wsum);
        lstm_gates      <<<64,   256, 0, stream>>>(emb_all, si, wsum, hs,
                                                   W_ih, W_hh, b_ih, b_hh, gates);
        lstm_update     <<<32,   128, 0, stream>>>(gates, hs, cs);
        pack_A          <<<70,   32,  0, stream>>>(hs, wsum, emb_all, si, pA);
        fc_gemm         <<<125,  256, 0, stream>>>(pA, pB, fc_b, out, t);
    }
    argmax_kernel<<<32 * 96, 256, 0, stream>>>(out, pred);
}
